// STSAMP_59219009077946
// MI455X (gfx1250) — matrix operand order repaired and checked by reading, not compiled, not run on hardware
//
#include <hip/hip_runtime.h>
#include <hip/hip_bf16.h>

// ---------------------------------------------------------------------------
// STSAMP fused forward for MI455X (gfx1250, wave32, WMMA).
//
// B=512, L=200, D=128, T=4.  One workgroup per batch row b.
// Never materializes [T,B,L,D] (840MB in the reference graph) — everything
// larger than [B,D] lives in LDS/registers.  emb (51MB) is L2-resident.
// ---------------------------------------------------------------------------

typedef __attribute__((ext_vector_type(16))) _Float16 v16h;
typedef __attribute__((ext_vector_type(8)))  float    v8f;

#define BATCH   512
#define LROWS   200
#define DDIM    128
#define TSTEPS  4
#define MTILES  13          // ceil(208/16) M-tiles (rows padded 200->208 with zeros)
#define EPITCH  132         // f32 LDS pitch for e rows: 132%64==4 -> bank spread
#define WPITCH  136         // f16 LDS pitch for w1 rows: 68 dwords, %64==4 -> spread

// LDS carve-up for kB (dynamic shared):
//   e_s   : 208*132 f32  = 109824 B   @ 0
//   w1h   : 128*136 f16  =  34816 B   @ 109824
//   vec_s : 4*128 f32    =   2048 B   @ 144640
//   seqS  : 208 int      =    832 B   @ 146688   (reused as outacc[128] f32)
#define OFF_W1H  109824
#define OFF_VEC  144640
#define OFF_SEQ  146688
#define SMEM_B   147520

// ---------------------------------------------------------------------------
// kernel 0: per-b mean-over-L and last-item row.  grid=B, block=128 (thread=d)
// ---------------------------------------------------------------------------
__global__ __launch_bounds__(128, 4)
void STSAMP_k0(const int* __restrict__ seq, const int* __restrict__ slen,
               const float* __restrict__ emb,
               float* __restrict__ ms_g, float* __restrict__ last_g)
{
    __shared__ int seqS[LROWS];
    const int b = blockIdx.x, tid = threadIdx.x;
    for (int i = tid; i < LROWS; i += 128) seqS[i] = seq[b * LROWS + i];
    __syncthreads();

    float s = 0.f;
    for (int l = 0; l < LROWS; ++l) {
        if (l + 4 < LROWS)  // CDNA5 global_prefetch_b8 path
            __builtin_prefetch(&emb[(size_t)seqS[l + 4] * DDIM + tid], 0, 3);
        s += emb[(size_t)seqS[l] * DDIM + tid];
    }
    const int len = slen[b];
    ms_g[b * DDIM + tid]   = s / (float)len;
    last_g[b * DDIM + tid] = emb[(size_t)seqS[len - 1] * DDIM + tid];
}

// ---------------------------------------------------------------------------
// block-wide sum over 128 threads (4 waves, wave32)
// ---------------------------------------------------------------------------
__device__ inline float bsum128(float v, float* red)
{
    #pragma unroll
    for (int m = 16; m >= 1; m >>= 1) v += __shfl_xor(v, m, 32);
    __syncthreads();
    if ((threadIdx.x & 31) == 0) red[threadIdx.x >> 5] = v;
    __syncthreads();
    return red[0] + red[1] + red[2] + red[3];
}

// ---------------------------------------------------------------------------
// kernel A: per-b small branches -> qv[t,b] scalars and ht[b,:].
// grid=B, block=128 (thread = output feature n)
// ---------------------------------------------------------------------------
__global__ __launch_bounds__(128, 4)
void STSAMP_kA(const float* __restrict__ w2, const float* __restrict__ w3,
               const float* __restrict__ w0, const float* __restrict__ ln2g,
               const float* __restrict__ ln2b, const float* __restrict__ ln3g,
               const float* __restrict__ ln3b, const float* __restrict__ mlp_b,
               const float* __restrict__ ms_g, const float* __restrict__ last_g,
               float* __restrict__ qv_g, float* __restrict__ ht_g)
{
    __shared__ float lastS[DDIM], msS[DDIM], red[4];
    const int b = blockIdx.x, tid = threadIdx.x;
    lastS[tid] = last_g[b * DDIM + tid];
    msS[tid]   = ms_g[b * DDIM + tid];
    __syncthreads();

    const float* w2r = &w2[tid * DDIM];
    const float* w3r = &w3[tid * DDIM];
    const float* mbr = &mlp_b[tid * DDIM];
    float y2 = 0.f, y3 = 0.f, yb = 0.f;
    for (int k = 0; k < DDIM; ++k) {
        y2 += lastS[k] * w2r[k];
        y3 += msS[k]   * w3r[k];
        yb += lastS[k] * mbr[k];
    }

    // LN(y2) with ln2, LN(y3) with ln3
    float s  = bsum128(y2, red);
    float sq = bsum128(y2 * y2, red);
    float mu = s * (1.f / DDIM);
    float vr = sq * (1.f / DDIM) - mu * mu;
    const float q = (y2 - mu) * rsqrtf(vr + 1e-5f) * ln2g[tid] + ln2b[tid];

    s  = bsum128(y3, red);
    sq = bsum128(y3 * y3, red);
    mu = s * (1.f / DDIM);
    vr = sq * (1.f / DDIM) - mu * mu;
    const float vv = (y3 - mu) * rsqrtf(vr + 1e-5f) * ln3g[tid] + ln3b[tid];

    // Q/V LIF over T (constant input), qv[t] = sum_n (sQ+sV)*w0[n]
    const float w0n = w0[tid];
    float vq = 0.f, vV = 0.f;
    for (int t = 0; t < TSTEPS; ++t) {
        float hq = 0.5f * (vq + q);
        float hv = 0.5f * (vV + vv);
        bool  sQ = (hq >= 1.f);
        bool  sV = (hv >= 1.f);
        vq = sQ ? 0.f : hq;
        vV = sV ? 0.f : hv;
        float contrib = ((sQ ? 1.f : 0.f) + (sV ? 1.f : 0.f)) * w0n;
        float tot = bsum128(contrib, red);
        if (tid == 0) qv_g[b * TSTEPS + t] = tot;
    }
    ht_g[b * DDIM + tid] = tanhf(yb);
}

// ---------------------------------------------------------------------------
// kernel B: fused  gather -> GEMM(e@w1^T, WMMA f16) -> LN -> K-LIF -> alpha
//           -> spike_alpha -> vec accumulation -> +ms -> tanh-MLP -> *ht -> out
// grid=B, block=256 (8 wave32s); dynamic LDS = SMEM_B (CDNA5 320KB/WG)
// ---------------------------------------------------------------------------
__global__ __launch_bounds__(256, 1)
void STSAMP_kB(const int* __restrict__ seq, const float* __restrict__ emb,
               const float* __restrict__ w1, const float* __restrict__ ln1g,
               const float* __restrict__ ln1b, const float* __restrict__ w0,
               const float* __restrict__ mlp_a, const float* __restrict__ ms_g,
               const float* __restrict__ qv_g, const float* __restrict__ ht_g,
               float* __restrict__ out)
{
    extern __shared__ char smem[];
    float*    e_s   = (float*)smem;                 // [208][EPITCH]
    _Float16* w1h   = (_Float16*)(smem + OFF_W1H);  // [128][WPITCH]
    float*    vec_s = (float*)(smem + OFF_VEC);     // [4][128]
    int*      seqS  = (int*)(smem + OFF_SEQ);       // [208], reused at end

    const int b    = blockIdx.x;
    const int tid  = threadIdx.x;
    const int lane = tid & 31;
    const int wv   = tid >> 5;
    const int r    = lane & 15;   // fragment row-within-halfwave
    const int g    = lane >> 4;   // halfwave select

    // ---- stage: seq, zero vec accumulators ----
    for (int i = tid; i < LROWS; i += 256) seqS[i] = seq[b * LROWS + i];
    for (int i = tid; i < TSTEPS * DDIM; i += 256) vec_s[i] = 0.f;
    __syncthreads();

    // ---- stage: gather e rows (padded to 208 with zeros), w1 -> f16 ----
    for (int idx = tid; idx < 208 * DDIM; idx += 256) {
        int l = idx >> 7, d = idx & 127;
        float v = 0.f;
        if (l < LROWS) v = emb[(size_t)seqS[l] * DDIM + d];
        e_s[l * EPITCH + d] = v;
    }
    for (int idx = tid; idx < DDIM * DDIM; idx += 256) {
        int n = idx >> 7, k = idx & 127;
        w1h[n * WPITCH + k] = (_Float16)w1[n * DDIM + k];
    }
    __syncthreads();

    // ---- per-lane constants: ln1 gamma/beta and w0, at n = nt*16 + r ----
    float gg[8], bb[8], w0v[8];
    #pragma unroll
    for (int nt = 0; nt < 8; ++nt) {
        int n = nt * 16 + r;
        gg[nt] = ln1g[n]; bb[nt] = ln1b[n]; w0v[nt] = w0[n];
    }
    float qv[TSTEPS];
    #pragma unroll
    for (int t = 0; t < TSTEPS; ++t) qv[t] = qv_g[b * TSTEPS + t];

    float vp[TSTEPS][8];   // per-lane vec partials at n = nt*16 + r
    #pragma unroll
    for (int t = 0; t < TSTEPS; ++t)
        #pragma unroll
        for (int nt = 0; nt < 8; ++nt) vp[t][nt] = 0.f;

    // ---- main loop: each wave owns M-tiles mt = wv, wv+8 ----
    for (int mt = wv; mt < MTILES; mt += 8) {
        v8f c[8];
        #pragma unroll
        for (int nt = 0; nt < 8; ++nt)
            c[nt] = (v8f){0.f, 0.f, 0.f, 0.f, 0.f, 0.f, 0.f, 0.f};

        // GEMM: y[16 x 128] = e_tile[16 x 128] @ w1^T, K in 4 steps of 32
        #pragma unroll
        for (int ks = 0; ks < 4; ++ks) {
            const int k0 = ks * 32;
            // A fragment (lane=m; contiguous K order chosen here: halfwave g holds
            // K = k0+16g+i, matching B)
            v16h a;
            const float* ep = &e_s[(mt * 16 + r) * EPITCH + k0 + 16 * g];
            #pragma unroll
            for (int i = 0; i < 8; ++i) a[i]     = (_Float16)ep[i];
            #pragma unroll
            for (int i = 0; i < 8; ++i) a[8 + i] = (_Float16)ep[8 + i];

            #pragma unroll
            for (int nt = 0; nt < 8; ++nt) {
                // B fragment (32x16, lane=n, halfwave g holds K = k0+16g+i).
                // B[k][n] = w1[n][k], so lane n streams w1 row n directly.
                v16h bf;
                const _Float16* wp = &w1h[(nt * 16 + r) * WPITCH + k0 + 16 * g];
                #pragma unroll
                for (int i = 0; i < 16; ++i) bf[i] = wp[i];
                c[nt] = __builtin_amdgcn_wmma_f32_16x16x32_f16(
                    false, a, false, bf, (short)0, c[nt], false, false);
            }
        }

        // ---- fused LN + K-LIF + w0-dot + spike_alpha + vec accumulation ----
        // C layout: lane holds y[m = j + 8g][n = nt*16 + r] in c[nt][j].
        const int rowBase = mt * 16 + 8 * g;
        #pragma unroll
        for (int j = 0; j < 8; ++j) {
            float s1 = 0.f, s2 = 0.f;
            #pragma unroll
            for (int nt = 0; nt < 8; ++nt) {
                float x = c[nt][j]; s1 += x; s2 += x * x;
            }
            #pragma unroll
            for (int m = 8; m >= 1; m >>= 1) {       // reduce within 16-lane half
                s1 += __shfl_xor(s1, m, 32);
                s2 += __shfl_xor(s2, m, 32);
            }
            const float mu   = s1 * (1.f / DDIM);
            const float var  = s2 * (1.f / DDIM) - mu * mu;
            const float rstd = rsqrtf(var + 1e-5f);

            float dA[TSTEPS] = {0.f, 0.f, 0.f, 0.f};
            #pragma unroll
            for (int nt = 0; nt < 8; ++nt) {
                const float xh = (c[nt][j] - mu) * rstd * gg[nt] + bb[nt];
                float v = 0.f;
                #pragma unroll
                for (int t = 0; t < TSTEPS; ++t) {   // K-LIF (constant input)
                    float h = 0.5f * (v + xh);
                    bool  sp = (h >= 1.f);
                    dA[t] += sp ? w0v[nt] : 0.f;
                    v = sp ? 0.f : h;
                }
            }
            #pragma unroll
            for (int t = 0; t < TSTEPS; ++t)
                #pragma unroll
                for (int m = 8; m >= 1; m >>= 1)
                    dA[t] += __shfl_xor(dA[t], m, 32);

            // spike_alpha LIF (time-varying input alpha[t] = dA[t] + qv[t])
            const int row = rowBase + j;
            float ev[8];
            #pragma unroll
            for (int nt = 0; nt < 8; ++nt)
                ev[nt] = e_s[row * EPITCH + nt * 16 + r];
            float va = 0.f;
            #pragma unroll
            for (int t = 0; t < TSTEPS; ++t) {
                float h  = 0.5f * (va + dA[t] + qv[t]);
                bool  sp = (h >= 1.f);
                va = sp ? 0.f : h;
                if (sp) {
                    #pragma unroll
                    for (int nt = 0; nt < 8; ++nt) vp[t][nt] += ev[nt];
                }
            }
        }
    }

    // ---- flush vec partials (ds_add_f32) ----
    #pragma unroll
    for (int t = 0; t < TSTEPS; ++t)
        #pragma unroll
        for (int nt = 0; nt < 8; ++nt)
            atomicAdd(&vec_s[t * DDIM + nt * 16 + r], vp[t][nt]);
    __syncthreads();

    // ---- ma = vec + ms (in place) ----
    for (int i = tid; i < TSTEPS * DDIM; i += 256)
        vec_s[i] += ms_g[b * DDIM + (i & 127)];
    __syncthreads();

    // ---- hs = tanh(ma @ mlp_a^T); out = 0.25 * sum_t hs * ht ----
    const int n     = tid & 127;
    const int thalf = tid >> 7;              // thread handles t = thalf, thalf+2
    const float* arow = &mlp_a[n * DDIM];
    float o = 0.f;
    #pragma unroll
    for (int tt = 0; tt < 2; ++tt) {
        const int t = thalf + 2 * tt;
        float acc = 0.f;
        for (int k = 0; k < DDIM; ++k) acc += vec_s[t * DDIM + k] * arow[k];
        o += tanhf(acc);
    }
    float* outacc = (float*)seqS;            // seqS dead; reuse as f32[128]
    if (thalf == 0) outacc[n] = o;
    __syncthreads();
    if (thalf == 1) outacc[n] += o;
    __syncthreads();
    if (thalf == 0)
        out[b * DDIM + n] = 0.25f * outacc[n] * ht_g[b * DDIM + n];
}

// ---------------------------------------------------------------------------
extern "C" void kernel_launch(void* const* d_in, const int* in_sizes, int n_in,
                              void* d_out, int out_size, void* d_ws, size_t ws_size,
                              hipStream_t stream)
{
    const int*   seq   = (const int*)  d_in[0];
    const int*   slen  = (const int*)  d_in[1];
    const float* emb   = (const float*)d_in[2];
    const float* w1    = (const float*)d_in[3];
    const float* w2    = (const float*)d_in[4];
    const float* w3    = (const float*)d_in[5];
    const float* w0    = (const float*)d_in[6];
    const float* ln1g  = (const float*)d_in[7];
    const float* ln1b  = (const float*)d_in[8];
    const float* ln2g  = (const float*)d_in[9];
    const float* ln2b  = (const float*)d_in[10];
    const float* ln3g  = (const float*)d_in[11];
    const float* ln3b  = (const float*)d_in[12];
    const float* mlpa  = (const float*)d_in[13];
    const float* mlpb  = (const float*)d_in[14];
    float*       out   = (float*)d_out;

    // workspace carve-up (deterministic)
    char* ws = (char*)d_ws;
    float* ms_g   = (float*)(ws);                          // B*D
    float* last_g = (float*)(ws + 262144);                 // B*D
    float* qv_g   = (float*)(ws + 524288);                 // T*B
    float* ht_g   = (float*)(ws + 532480);                 // B*D

    STSAMP_k0<<<BATCH, 128, 0, stream>>>(seq, slen, emb, ms_g, last_g);

    STSAMP_kA<<<BATCH, 128, 0, stream>>>(w2, w3, w0, ln2g, ln2b, ln3g, ln3b,
                                         mlpb, ms_g, last_g, qv_g, ht_g);

    // CDNA5: 320KB LDS per workgroup; request large dynamic LDS.
    (void)hipFuncSetAttribute((const void*)STSAMP_kB,
                              hipFuncAttributeMaxDynamicSharedMemorySize,
                              SMEM_B);
    STSAMP_kB<<<BATCH, 256, SMEM_B, stream>>>(seq, emb, w1, ln1g, ln1b, w0,
                                              mlpa, ms_g, qv_g, ht_g, out);
}
